// SOM_50294067036419
// MI455X (gfx1250) — matrix operand order repaired and checked by reading, not compiled, not run on hardware
//
#include <hip/hip_runtime.h>
#include <cstdint>

#define T_DIM   1024
#define K_DIM   256
#define NTILE_T (T_DIM / 16)   // 64 weight row-tiles

typedef __attribute__((ext_vector_type(16))) __bf16   v16bf;
typedef __attribute__((ext_vector_type(8)))  float    v8f;
typedef __attribute__((ext_vector_type(4)))  float    f32x4;
typedef __attribute__((ext_vector_type(4)))  unsigned u32x4;

union V16 {            // 8 VGPRs viewed as bf16x16 / 2x128b / 8x u32
  u32x4   q[2];
  unsigned u[8];
  v16bf   v;
};

static __device__ __forceinline__ unsigned f2bf1(float f) {
  // round-to-nearest-even f32 -> bf16 (bits in [15:0])
  unsigned u = __float_as_uint(f);
  return (u + 0x7fffu + ((u >> 16) & 1u)) >> 16;
}
static __device__ __forceinline__ unsigned packbf(float lo, float hi) {
  return f2bf1(lo) | (f2bf1(hi) << 16);
}

// ---------------------------------------------------------------------------
// Kernel 0: W (f32) -> bf16 copy in ws, ||w||^2 per row, zero BMU mask.
// ---------------------------------------------------------------------------
__global__ void som_prep(const float* __restrict__ W,
                         unsigned short* __restrict__ wbf,
                         float* __restrict__ w2,
                         unsigned* __restrict__ mask) {
  int t = blockIdx.x * blockDim.x + threadIdx.x;
  if (t >= T_DIM) return;
  const float* row = W + (size_t)t * K_DIM;
  unsigned short* orow = wbf + (size_t)t * K_DIM;
  float s = 0.f;
#pragma unroll 4
  for (int k = 0; k < K_DIM; ++k) {
    float v = row[k];
    s += v * v;
    orow[k] = (unsigned short)f2bf1(v);
  }
  w2[t] = s;
  mask[t] = 0u;
}

// ---------------------------------------------------------------------------
// Kernel 1: BMU search. Each wave: one 16-column data tile vs all 1024 rows.
// D-tile = W_bf16 x X_bf16 with C pre-biased by -0.5*||w||^2, BMU = argmax.
// ---------------------------------------------------------------------------
__global__ void __launch_bounds__(128)
som_bmu(const float* __restrict__ X,
        const unsigned short* __restrict__ wbf,
        const float* __restrict__ w2,
        unsigned* __restrict__ mask,
        int ndata) {
  const int lane    = threadIdx.x & 31;
  const int wave    = threadIdx.x >> 5;
  const int colBase = blockIdx.x * 64 + wave * 16;
  const int col     = colBase + (lane & 15);
  const int colc    = (col < ndata) ? col : (ndata - 1);   // clamp tail (select, no branch)
  const int halfSel = lane >> 4;                           // 0: lanes 0-15, 1: lanes 16-31

  // ---- Stage B operand (this wave's 16 data columns) in registers ---------
  // B 32x16 bf16 layout: lane n<16 -> col n, K 0..15; lanes 16..31 -> K 16..31.
  const float* drow = X + (size_t)colc * K_DIM;
  const int kB = halfSel * 16;
  v16bf b[8];
#pragma unroll
  for (int kc = 0; kc < 8; ++kc) {
    const float* p = drow + kc * 32 + kB;
    f32x4 d0 = *(const f32x4*)(p);
    f32x4 d1 = *(const f32x4*)(p + 4);
    f32x4 d2 = *(const f32x4*)(p + 8);
    f32x4 d3 = *(const f32x4*)(p + 12);
    V16 bv;
    bv.u[0] = packbf(d0.x, d0.y); bv.u[1] = packbf(d0.z, d0.w);
    bv.u[2] = packbf(d1.x, d1.y); bv.u[3] = packbf(d1.z, d1.w);
    bv.u[4] = packbf(d2.x, d2.y); bv.u[5] = packbf(d2.z, d2.w);
    bv.u[6] = packbf(d3.x, d3.y); bv.u[7] = packbf(d3.z, d3.w);
    b[kc] = bv.v;
  }

  // A 16x32 bf16 layout: lane<16 -> K 0..15; lane>=16 -> K 16..31
  const int kA   = halfSel * 16;
  const int mOff = halfSel * 8;   // C: VGPR r holds M=r (lanes 0-15) / M=r+8 (16-31)

  float bestm  = -3.402823466e38f;
  int   bestrow = 0;

  for (int tile = 0; tile < NTILE_T; ++tile) {
    const unsigned short* wrow =
        wbf + (size_t)(tile * 16 + (lane & 15)) * K_DIM;
    if (tile + 1 < NTILE_T)   // gfx1250 global_prefetch_b8 on next W tile
      __builtin_prefetch(wbf + (size_t)((tile + 1) * 16 + (lane & 15)) * K_DIM, 0, 3);

    // Accumulator pre-biased with -0.5*||w||^2 -> D = w.x - 0.5||w||^2
    f32x4 wa = *(const f32x4*)(w2 + tile * 16 + mOff);
    f32x4 wb = *(const f32x4*)(w2 + tile * 16 + mOff + 4);
    v8f c;
    c[0] = -0.5f * wa.x; c[1] = -0.5f * wa.y;
    c[2] = -0.5f * wa.z; c[3] = -0.5f * wa.w;
    c[4] = -0.5f * wb.x; c[5] = -0.5f * wb.y;
    c[6] = -0.5f * wb.z; c[7] = -0.5f * wb.w;

#pragma unroll
    for (int kc = 0; kc < 8; ++kc) {
      V16 av;
      av.q[0] = *(const u32x4*)(wrow + kc * 32 + kA);        // K .. +7   (or +16..+23)
      av.q[1] = *(const u32x4*)(wrow + kc * 32 + 8 + kA);    // K+8..+15  (or +24..+31)
      c = __builtin_amdgcn_wmma_f32_16x16x32_bf16(
              /*neg_a=*/false, av.v, /*neg_b=*/false, b[kc],
              /*c_mod=*/(short)0, c, /*reuse_a=*/false, /*reuse_b=*/false);
    }

    // per-lane argmax over the 8 rows this lane holds
    float m = c[0]; int mi = 0;
#pragma unroll
    for (int r = 1; r < 8; ++r) {
      if (c[r] > m) { m = c[r]; mi = r; }
    }
    int row = tile * 16 + mOff + mi;
    if (m > bestm) { bestm = m; bestrow = row; }
  }

  // lane n and lane n^16 hold the same data column (rows M and M+8): merge.
  float om   = __shfl_xor(bestm, 16, 32);
  int   orow = __shfl_xor(bestrow, 16, 32);
  if (om > bestm || (om == bestm && orow < bestrow)) { bestm = om; bestrow = orow; }

  if (lane < 16 && col < ndata) mask[bestrow] = 1u;   // same-value stores: race-free
}

// ---------------------------------------------------------------------------
// Kernel 2: out = mask[t] ? W : 0
// ---------------------------------------------------------------------------
__global__ void som_writeout(const float* __restrict__ W,
                             const unsigned* __restrict__ mask,
                             float* __restrict__ out) {
  int i = blockIdx.x * blockDim.x + threadIdx.x;
  if (i >= T_DIM * K_DIM) return;
  out[i] = mask[i >> 8] ? W[i] : 0.0f;
}

// ---------------------------------------------------------------------------
extern "C" void kernel_launch(void* const* d_in, const int* in_sizes, int n_in,
                              void* d_out, int out_size, void* d_ws, size_t ws_size,
                              hipStream_t stream) {
  const float* W = (const float*)d_in[0];   // [1024, 256] f32
  const float* X = (const float*)d_in[1];   // [N, 256] f32
  float* out = (float*)d_out;               // [1024, 256] f32

  char* ws = (char*)d_ws;
  unsigned*       mask = (unsigned*)ws;                  //   4 KB
  float*          w2   = (float*)(ws + 4096);            //   4 KB
  unsigned short* wbf  = (unsigned short*)(ws + 8192);   // 512 KB bf16 W

  int ndata = in_sizes[1] / K_DIM;

  som_prep<<<(T_DIM + 255) / 256, 256, 0, stream>>>(W, wbf, w2, mask);
  som_bmu<<<(ndata + 63) / 64, 128, 0, stream>>>(X, wbf, w2, mask, ndata);
  som_writeout<<<(T_DIM * K_DIM + 255) / 256, 256, 0, stream>>>(W, mask, out);
}
